// CrossAttention_37366215475281
// MI455X (gfx1250) — matrix operand order repaired and checked by reading, not compiled, not run on hardware
//
#include <hip/hip_runtime.h>

// MI455X / gfx1250, wave32. All matmuls via v_wmma_f32_16x16x32_bf16.
// Pipeline: qkv_proj -> scores GEMM -> row softmax (in-place f32->bf16) -> PV GEMM.
// Workspace layout (bytes): Q[0,32M) K[32M,64M) V^T[64M,96M) scores/P[96M,224M)
// GEMMs: 128x128x32 block tiles, 8 waves, double-buffered LDS + register staging.

typedef __attribute__((ext_vector_type(16))) __bf16 bf16x16;
typedef __attribute__((ext_vector_type(8)))  float  v8f;

#define HD   1024
#define SEQ  2048
#define NBAT 8
#define BM   128
#define BN   128
#define BK   32
#define LDP  48   // padded LDS row stride in bf16 elements (96B, 16B-aligned)

union FragU { bf16x16 v; uint4 q[2]; };

__device__ inline bf16x16 load_frag2(const __bf16* p0, const __bf16* p1) {
    FragU f;
    f.q[0] = *(const uint4*)p0;
    f.q[1] = *(const uint4*)p1;
    return f.v;
}

__device__ inline v8f wmma_bf16(bf16x16 a, bf16x16 b, v8f c) {
    return __builtin_amdgcn_wmma_f32_16x16x32_bf16(
        /*neg_a=*/false, a, /*neg_b=*/false, b,
        /*c_mod=*/(short)0, c, /*reuse_a=*/false, /*reuse_b=*/false);
}

// ---------------------------------------------------------------------------
// Kernel 1: QKV projection.  out = x @ W + bias, bf16 output.
// grid = (HD/BN, (NBAT*SEQ)/BM, 3); z selects (Wq,bq,Q),(Wk,bk,K),(Wv,bv,V^T)
// x: [16384,1024] f32 row-major.  W: [1024,1024] f32 row-major (k-major).
// ---------------------------------------------------------------------------
__global__ __launch_bounds__(256)
void qkv_proj(const float* __restrict__ x,
              const float* __restrict__ W0, const float* __restrict__ W1,
              const float* __restrict__ W2,
              const float* __restrict__ b0, const float* __restrict__ b1,
              const float* __restrict__ b2,
              __bf16* __restrict__ Qo, __bf16* __restrict__ Ko,
              __bf16* __restrict__ VTo) {
    __shared__ __bf16 sA[2][BM * LDP];
    __shared__ __bf16 sB[2][BN * LDP];

    const int tid  = threadIdx.x;
    const int lane = tid & 31;
    const int wave = tid >> 5;
    const int wm0  = (wave >> 1) * 32;   // 4 waves along M
    const int wn0  = (wave & 1) * 64;    // 2 waves along N
    const long m0  = (long)blockIdx.y * BM;
    const long n0  = (long)blockIdx.x * BN;
    const int which = blockIdx.z;
    const float* W    = (which == 0) ? W0 : ((which == 1) ? W1 : W2);
    const float* bias = (which == 0) ? b0 : ((which == 1) ? b1 : b2);

    const int ln16  = lane & 15;
    const int hi    = lane >> 4;
    const int abase = hi ? 8 : 0;    // A-frag k offsets: {abase..+7, abase+16..+23}
    const int bbase = hi ? 8 : 0;    // B-frag k offsets: {bbase..+7, bbase+16..+23}

    // loader coordinates (constant across k-steps)
    const int rA = tid >> 3;          // + 32*i ; 128 rows
    const int cA = (tid & 7) * 4;     // f32 col
    const int kB = tid >> 5;          // + 8*i  ; 32 k-rows
    const int nB = (tid & 31) * 4;    // n col

    v8f zero = {0.f, 0.f, 0.f, 0.f, 0.f, 0.f, 0.f, 0.f};
    v8f acc[2][4];
#pragma unroll
    for (int i = 0; i < 2; i++)
#pragma unroll
        for (int j = 0; j < 4; j++) acc[i][j] = zero;

    // ---- preload k0 = 0 into buffer 0
#pragma unroll
    for (int i = 0; i < 4; i++) {
        float4 v = *(const float4*)&x[(m0 + rA + 32 * i) * (long)HD + cA];
        union { __bf16 h[4]; unsigned long long u; } pk;
        pk.h[0] = (__bf16)v.x; pk.h[1] = (__bf16)v.y;
        pk.h[2] = (__bf16)v.z; pk.h[3] = (__bf16)v.w;
        *(unsigned long long*)&sA[0][(rA + 32 * i) * LDP + cA] = pk.u;

        float4 w = *(const float4*)&W[(long)(kB + 8 * i) * HD + n0 + nB];
        sB[0][(nB + 0) * LDP + kB + 8 * i] = (__bf16)w.x;
        sB[0][(nB + 1) * LDP + kB + 8 * i] = (__bf16)w.y;
        sB[0][(nB + 2) * LDP + kB + 8 * i] = (__bf16)w.z;
        sB[0][(nB + 3) * LDP + kB + 8 * i] = (__bf16)w.w;
    }
    __syncthreads();

    int buf = 0;
    for (int k0 = 0; k0 < HD; k0 += BK) {
        const bool nxt = (k0 + BK) < HD;
        float4 ra[4], rb[4];
        if (nxt) {
#pragma unroll
            for (int i = 0; i < 4; i++) {
                ra[i] = *(const float4*)&x[(m0 + rA + 32 * i) * (long)HD + (k0 + BK) + cA];
                rb[i] = *(const float4*)&W[(long)(k0 + BK + kB + 8 * i) * HD + n0 + nB];
            }
        }

        const __bf16* cubA = sA[buf];
        const __bf16* cubB = sB[buf];
        bf16x16 af[2], bf[4];
#pragma unroll
        for (int im = 0; im < 2; im++) {
            const __bf16* p = &cubA[(wm0 + im * 16 + ln16) * LDP + abase];
            af[im] = load_frag2(p, p + 16);
        }
#pragma unroll
        for (int in = 0; in < 4; in++) {
            const __bf16* p = &cubB[(wn0 + in * 16 + ln16) * LDP + bbase];
            bf[in] = load_frag2(p, p + 16);
        }
#pragma unroll
        for (int im = 0; im < 2; im++)
#pragma unroll
            for (int in = 0; in < 4; in++)
                acc[im][in] = wmma_bf16(af[im], bf[in], acc[im][in]);

        if (nxt) {
            __bf16* nxA = sA[buf ^ 1];
            __bf16* nxB = sB[buf ^ 1];
#pragma unroll
            for (int i = 0; i < 4; i++) {
                union { __bf16 h[4]; unsigned long long u; } pk;
                pk.h[0] = (__bf16)ra[i].x; pk.h[1] = (__bf16)ra[i].y;
                pk.h[2] = (__bf16)ra[i].z; pk.h[3] = (__bf16)ra[i].w;
                *(unsigned long long*)&nxA[(rA + 32 * i) * LDP + cA] = pk.u;
                nxB[(nB + 0) * LDP + kB + 8 * i] = (__bf16)rb[i].x;
                nxB[(nB + 1) * LDP + kB + 8 * i] = (__bf16)rb[i].y;
                nxB[(nB + 2) * LDP + kB + 8 * i] = (__bf16)rb[i].z;
                nxB[(nB + 3) * LDP + kB + 8 * i] = (__bf16)rb[i].w;
            }
        }
        __syncthreads();
        buf ^= 1;
    }

    // ---- Epilogue: bias add + bf16 store; branch hoisted (uniform per block)
    const int mrow0 = hi ? 8 : 0;
    if (which < 2) {
        __bf16* __restrict__ O = which ? Ko : Qo;
#pragma unroll
        for (int im = 0; im < 2; im++) {
#pragma unroll
            for (int in = 0; in < 4; in++) {
                long n = n0 + wn0 + in * 16 + ln16;
                float bv = bias[n];
#pragma unroll
                for (int j = 0; j < 8; j++) {
                    long m = m0 + wm0 + im * 16 + mrow0 + j;
                    O[m * (long)HD + n] = (__bf16)(acc[im][in][j] + bv);
                }
            }
        }
    } else {
        const long bb = m0 >> 11;                 // batch (128 | 2048)
#pragma unroll
        for (int im = 0; im < 2; im++) {
            const long s0 = (m0 + wm0 + im * 16 + mrow0) & (SEQ - 1);
#pragma unroll
            for (int in = 0; in < 4; in++) {
                long n = n0 + wn0 + in * 16 + ln16;
                float bv = bias[n];
                union { __bf16 h[8]; uint4 q; } pk;
#pragma unroll
                for (int j = 0; j < 8; j++)
                    pk.h[j] = (__bf16)(acc[im][in][j] + bv);
                // 8 consecutive tokens at [bb][n][s0..s0+7] -> one 16B store
                *(uint4*)&VTo[bb * (long)HD * SEQ + n * (long)SEQ + s0] = pk.q;
            }
        }
    }
}

// ---------------------------------------------------------------------------
// Kernel 2/4: generic bf16 GEMM, B given n-major ([n][k] row-major), f32 out.
// C[b] = scale * A[b] x B[b].  grid = (N/BN, M/BM, batches)
// ---------------------------------------------------------------------------
__global__ __launch_bounds__(256)
void gemm_bf16_nmajor(const __bf16* __restrict__ A, long lda, long aBS,
                      const __bf16* __restrict__ Bm, long ldb, long bBS,
                      float* __restrict__ C, long ldo, long cBS,
                      int Kdim, float scale) {
    __shared__ __bf16 sA[2][BM * LDP];
    __shared__ __bf16 sB[2][BN * LDP];

    const int tid  = threadIdx.x;
    const int lane = tid & 31;
    const int wave = tid >> 5;
    const int wm0  = (wave >> 1) * 32;
    const int wn0  = (wave & 1) * 64;
    const long m0  = (long)blockIdx.y * BM;
    const long n0  = (long)blockIdx.x * BN;
    A  += (long)blockIdx.z * aBS;
    Bm += (long)blockIdx.z * bBS;
    C  += (long)blockIdx.z * cBS;

    const int ln16  = lane & 15;
    const int hi    = lane >> 4;
    const int abase = hi ? 8 : 0;
    const int bbase = hi ? 8 : 0;

    const int r0 = tid >> 2;          // + 64*i ; 128 rows
    const int c0 = (tid & 3) * 8;     // bf16 col

    v8f zero = {0.f, 0.f, 0.f, 0.f, 0.f, 0.f, 0.f, 0.f};
    v8f acc[2][4];
#pragma unroll
    for (int i = 0; i < 2; i++)
#pragma unroll
        for (int j = 0; j < 4; j++) acc[i][j] = zero;

    // ---- preload k0 = 0 into buffer 0
#pragma unroll
    for (int i = 0; i < 2; i++) {
        *(uint4*)&sA[0][(r0 + 64 * i) * LDP + c0] =
            *(const uint4*)&A[(m0 + r0 + 64 * i) * lda + c0];
        *(uint4*)&sB[0][(r0 + 64 * i) * LDP + c0] =
            *(const uint4*)&Bm[(n0 + r0 + 64 * i) * ldb + c0];
    }
    __syncthreads();

    int buf = 0;
    for (int k0 = 0; k0 < Kdim; k0 += BK) {
        const bool nxt = (k0 + BK) < Kdim;
        uint4 ra[2], rb[2];
        if (nxt) {
#pragma unroll
            for (int i = 0; i < 2; i++) {
                ra[i] = *(const uint4*)&A[(m0 + r0 + 64 * i) * lda + (k0 + BK) + c0];
                rb[i] = *(const uint4*)&Bm[(n0 + r0 + 64 * i) * ldb + (k0 + BK) + c0];
            }
        }

        const __bf16* cubA = sA[buf];
        const __bf16* cubB = sB[buf];
        bf16x16 af[2], bf[4];
#pragma unroll
        for (int im = 0; im < 2; im++) {
            const __bf16* p = &cubA[(wm0 + im * 16 + ln16) * LDP + abase];
            af[im] = load_frag2(p, p + 16);
        }
#pragma unroll
        for (int in = 0; in < 4; in++) {
            const __bf16* p = &cubB[(wn0 + in * 16 + ln16) * LDP + bbase];
            bf[in] = load_frag2(p, p + 16);
        }
#pragma unroll
        for (int im = 0; im < 2; im++)
#pragma unroll
            for (int in = 0; in < 4; in++)
                acc[im][in] = wmma_bf16(af[im], bf[in], acc[im][in]);

        if (nxt) {
            __bf16* nxA = sA[buf ^ 1];
            __bf16* nxB = sB[buf ^ 1];
#pragma unroll
            for (int i = 0; i < 2; i++) {
                *(uint4*)&nxA[(r0 + 64 * i) * LDP + c0] = ra[i];
                *(uint4*)&nxB[(r0 + 64 * i) * LDP + c0] = rb[i];
            }
        }
        __syncthreads();
        buf ^= 1;
    }

    const int mrow0 = hi ? 8 : 0;
#pragma unroll
    for (int im = 0; im < 2; im++) {
#pragma unroll
        for (int in = 0; in < 4; in++) {
            long n = n0 + wn0 + in * 16 + ln16;
#pragma unroll
            for (int j = 0; j < 8; j++) {
                long m = m0 + wm0 + im * 16 + mrow0 + j;
                C[m * ldo + n] = acc[im][in][j] * scale;
            }
        }
    }
}

// ---------------------------------------------------------------------------
// Kernel 3: row softmax over 2048-wide rows of f32 scores; writes normalized
// probabilities as bf16 packed in-place at the start of each row's storage.
// One wave per row; row held in 64 VGPRs.
// ---------------------------------------------------------------------------
__global__ __launch_bounds__(256)
void softmax_rows(float* __restrict__ S) {
    const int lane = threadIdx.x & 31;
    const int wave = threadIdx.x >> 5;
    const long row = (long)blockIdx.x * 8 + wave;
    float* srow = S + row * (long)SEQ;

    float v[64];
    float m = -3.4e38f;
#pragma unroll
    for (int i = 0; i < 32; i++) {
        float2 t = *(const float2*)&srow[i * 64 + lane * 2];
        v[2 * i] = t.x;
        v[2 * i + 1] = t.y;
        m = fmaxf(m, fmaxf(t.x, t.y));
    }
#pragma unroll
    for (int o = 16; o > 0; o >>= 1) m = fmaxf(m, __shfl_xor(m, o, 32));

    float sum = 0.f;
#pragma unroll
    for (int i = 0; i < 64; i++) {
        v[i] = __expf(v[i] - m);
        sum += v[i];
    }
#pragma unroll
    for (int o = 16; o > 0; o >>= 1) sum += __shfl_xor(sum, o, 32);
    float inv = 1.0f / sum;

    unsigned* prow = (unsigned*)srow;   // bf16-pair packed, row stride 4096 bf16
#pragma unroll
    for (int i = 0; i < 32; i++) {
        union { __bf16 h[2]; unsigned u; } pk;
        pk.h[0] = (__bf16)(v[2 * i] * inv);
        pk.h[1] = (__bf16)(v[2 * i + 1] * inv);
        prow[i * 32 + lane] = pk.u;
    }
}

// ---------------------------------------------------------------------------
extern "C" void kernel_launch(void* const* d_in, const int* in_sizes, int n_in,
                              void* d_out, int out_size, void* d_ws, size_t ws_size,
                              hipStream_t stream) {
    const float* x  = (const float*)d_in[0];
    const float* Wq = (const float*)d_in[1];
    const float* bq = (const float*)d_in[2];
    const float* Wk = (const float*)d_in[3];
    const float* bk = (const float*)d_in[4];
    const float* Wv = (const float*)d_in[5];
    const float* bv = (const float*)d_in[6];
    float* out = (float*)d_out;

    char* ws = (char*)d_ws;              // needs >= 224 MB
    __bf16* Q  = (__bf16*)(ws);
    __bf16* Kb = (__bf16*)(ws + (size_t)32 * 1024 * 1024);
    __bf16* VT = (__bf16*)(ws + (size_t)64 * 1024 * 1024);
    float*  S  = (float*) (ws + (size_t)96 * 1024 * 1024);

    dim3 blk(256);

    // 1) Q,K (row-major) and V^T projections, bf16
    qkv_proj<<<dim3(HD / BN, (NBAT * SEQ) / BM, 3), blk, 0, stream>>>(
        x, Wq, Wk, Wv, bq, bk, bv, Q, Kb, VT);

    // 2) scores = Q @ K^T / sqrt(1024)   (K row-major [j][d] is n-major here)
    gemm_bf16_nmajor<<<dim3(SEQ / BN, SEQ / BM, NBAT), blk, 0, stream>>>(
        Q, (long)HD, (long)SEQ * HD,
        Kb, (long)HD, (long)SEQ * HD,
        S, (long)SEQ, (long)SEQ * SEQ,
        HD, 0.03125f);

    // 3) softmax rows, in-place f32 -> packed bf16 probabilities
    softmax_rows<<<dim3((NBAT * SEQ) / 8), blk, 0, stream>>>(S);

    // 4) out = P @ V   (P: bf16, row stride 4096 elems; V^T is n-major)
    gemm_bf16_nmajor<<<dim3(HD / BN, SEQ / BM, NBAT), blk, 0, stream>>>(
        (const __bf16*)S, (long)(2 * SEQ), (long)SEQ * 2 * SEQ,
        VT, (long)SEQ, (long)HD * SEQ,
        out, (long)HD, (long)SEQ * HD,
        SEQ, 1.0f);
}
